// SelfAttention_7670811591149
// MI455X (gfx1250) — matrix operand order repaired and checked by reading, not compiled, not run on hardware
//
#include <hip/hip_runtime.h>

// ---------- types ----------
typedef __bf16 bf16x16 __attribute__((ext_vector_type(16)));
typedef __bf16 bf16x2  __attribute__((ext_vector_type(2)));
typedef float  f32x8   __attribute__((ext_vector_type(8)));

// LDS tile row: 32 bf16 (64B data) padded to 80B -> 16B-aligned rows for B128
// async stores, 20-dword bank stride (conflict-free for 16 consecutive rows).
#define TROW_ELEMS 40
#define TROW_BYTES 80

// D = A(16x32 bf16) * B(32x16 bf16) + C(16x16 f32)
__device__ __forceinline__ f32x8 wmma_bf16(bf16x16 a, bf16x16 b, f32x8 c) {
    return __builtin_amdgcn_wmma_f32_16x16x32_bf16(
        false, a, false, b, (short)0, c, false, false);
}

// A fragment (16 rows M x 32 cols K), source row-major [M][stride] bf16.
// ISA 7.12.2: lanes 0-15 & 16-31 both M=lane%16;
// VGPR v: k0 = 2*(v%4) + 8*half + 16*(v/4), pair (k0,k0+1).
__device__ __forceinline__ bf16x16 load_a_frag(const __bf16* base, int stride) {
    int lane = threadIdx.x & 31;
    int half = (lane >> 4) & 1;
    int m = lane & 15;
    bf16x16 r;
#pragma unroll
    for (int v = 0; v < 8; ++v) {
        int k0 = 2 * (v & 3) + 8 * half + 16 * (v >> 2);
        bf16x2 e = *reinterpret_cast<const bf16x2*>(base + (size_t)m * stride + k0);
        r[2 * v] = e[0];
        r[2 * v + 1] = e[1];
    }
    return r;
}

// B fragment (32 rows K x 16 cols N), source stored N-major [N][stride] bf16
// (element (k,n) at base[n*stride + k]).
// Layout: lane n=lane%16; VGPR v: k0 = 2*(v%4) + 8*half + 16*(v/4), pair (k0,k0+1).
__device__ __forceinline__ bf16x16 load_b_frag(const __bf16* base, int stride) {
    int lane = threadIdx.x & 31;
    int half = (lane >> 4) & 1;
    int n = lane & 15;
    bf16x16 r;
#pragma unroll
    for (int v = 0; v < 8; ++v) {
        int k0 = 2 * (v & 3) + 8 * half + 16 * (v >> 2);
        bf16x2 e = *reinterpret_cast<const bf16x2*>(base + (size_t)n * stride + k0);
        r[2 * v] = e[0];
        r[2 * v + 1] = e[1];
    }
    return r;
}

// Async-stage one 64-row x 64-byte tile (rows at row_stride bytes in global)
// into LDS rows of 80 bytes, one B128 per thread (256 threads).
// ldsbuf passed as operand so the compiler knows the shared buffer is written.
__device__ __forceinline__ void async_stage_tile(const __bf16* gbase, int row_stride_bytes,
                                                 __bf16* ldsbuf) {
    int tid = threadIdx.x;
    int nl = tid >> 2, seg = tid & 3;
    unsigned ldsoff = (unsigned)(nl * TROW_BYTES + seg * 16);
    unsigned goff = (unsigned)(nl * row_stride_bytes + seg * 16);
    asm volatile("global_load_async_to_lds_b128 %0, %1, %2 ; lds buffer %3"
                 :: "v"(ldsoff), "v"(goff),
                    "s"((unsigned long long)(uintptr_t)gbase), "v"(ldsbuf)
                 : "memory");
    asm volatile("s_wait_asynccnt 0x0" ::: "memory");
}

// tcat[b, c, n] gather: n over (56 x 112), right half comes from batch b+2.
__device__ __forceinline__ float tcat_load(const float* t, int b, int c, int n) {
    int h = n / 112, ww = n % 112;
    int bb = b + (ww >= 56 ? 2 : 0);
    int w = ww >= 56 ? ww - 56 : ww;
    return t[((size_t)(bb * 256 + c)) * 3136 + h * 56 + w];
}

// ---------- kernel 1: fold BN into weights, convert to bf16 ----------
__global__ void pack_weights(const float* Wq, const float* Wk, const float* Wv,
                             const float* gv, const float* bv, const float* mv, const float* vv,
                             const float* Wo, const float* go, const float* bo, const float* mo, const float* vvo,
                             __bf16* wq_bf, __bf16* wk_bf, __bf16* wv_bf, __bf16* wo_bf,
                             float* bias_v, float* bias_o) {
    int i = blockIdx.x * blockDim.x + threadIdx.x;
    if (i < 64 * 256) {
        wq_bf[i] = (__bf16)Wq[i];
        wk_bf[i] = (__bf16)Wk[i];
    }
    if (i < 256 * 256) {
        int oc = i / 256;
        float inv = gv[oc] * rsqrtf(vv[oc] + 1e-5f);
        wv_bf[i] = (__bf16)(Wv[i] * inv);
    }
    if (i < 256 * 512) {
        int oc = i / 512;
        float inv = go[oc] * rsqrtf(vvo[oc] + 1e-5f);
        wo_bf[i] = (__bf16)(Wo[i] * inv);
    }
    if (i < 256) {
        float iv = gv[i] * rsqrtf(vv[i] + 1e-5f);
        bias_v[i] = bv[i] - mv[i] * iv;
        float io = go[i] * rsqrtf(vvo[i] + 1e-5f);
        bias_o[i] = bo[i] - mo[i] * io;
    }
}

// ---------- kernel 1b: Xn[b][n][256] bf16 = tcat (n-major) ----------
__global__ void pack_x(const float* t, __bf16* Xn) {
    int i = blockIdx.x * 256 + threadIdx.x;   // (b*6272+n)*256 + c
    int c = i & 255;
    int bn = i >> 8;
    int n = bn % 6272;
    int b = bn / 6272;
    Xn[i] = (__bf16)tcat_load(t, b, c, n);
}

// ---------- kernel 1c: On[bb][p][512] bf16 = concat(t, vo) (p-major) ----------
__global__ void pack_o(const float* t, const float* vo, __bf16* On) {
    int i = blockIdx.x * 256 + threadIdx.x;   // (bb*3136+p)*512 + ic
    int ic = i & 511;
    int bp = i >> 9;
    int p = bp % 3136;
    int bb = bp / 3136;
    float val;
    if (ic < 256) {
        val = t[((size_t)bb * 256 + ic) * 3136 + p];
    } else {
        int b = bb & 1, woff = (bb >= 2) ? 56 : 0;
        int h = p / 56, w = p % 56;
        val = vo[((size_t)b * 256 + (ic - 256)) * 6272 + h * 112 + w + woff];
    }
    On[i] = (__bf16)val;
}

// ---------- kernel 2: Q,K projections -> qT,kT [b][n][64] bf16 ----------
// 8 waves: waves 0-3 -> Q c-tiles, waves 4-7 -> K c-tiles; async-staged LDS tile.
__global__ void __launch_bounds__(256) qk_proj(const __bf16* Xn, const __bf16* wq, const __bf16* wk,
                                               __bf16* qT, __bf16* kT) {
    __shared__ __bf16 Bt[64 * TROW_ELEMS];
    int wg = blockIdx.x;          // 2 * 98
    int b = wg / 98;
    int n0 = (wg % 98) * 64;
    int tid = threadIdx.x, wave = tid >> 5, lane = tid & 31;
    const __bf16* Wmat = (wave < 4) ? wq : wk;
    int cq0 = (wave & 3) * 16;
    const __bf16* xrows = Xn + ((size_t)b * 6272 + n0) * 256;
    f32x8 acc[4] = {};
    for (int k0 = 0; k0 < 256; k0 += 32) {
        __syncthreads();
        async_stage_tile(xrows + k0, 256 * 2, Bt);
        __syncthreads();
        bf16x16 a = load_a_frag(Wmat + (size_t)cq0 * 256 + k0, 256);
#pragma unroll
        for (int nt = 0; nt < 4; ++nt) {
            bf16x16 bf = load_b_frag(Bt + nt * 16 * TROW_ELEMS, TROW_ELEMS);
            acc[nt] = wmma_bf16(a, bf, acc[nt]);
        }
    }
    __bf16* outp = ((wave < 4) ? qT : kT) + (size_t)b * 6272 * 64;
#pragma unroll
    for (int nt = 0; nt < 4; ++nt)
#pragma unroll
        for (int v = 0; v < 8; ++v) {
            int n = n0 + nt * 16 + (lane & 15);
            int cq = cq0 + v + 8 * ((lane >> 4) & 1);
            outp[(size_t)n * 64 + cq] = (__bf16)acc[nt][v];
        }
}

// ---------- kernel 3: V projection + BN/ReLU -> v [b][c][n] bf16 ----------
__global__ void __launch_bounds__(256) v_proj(const __bf16* Xn, const __bf16* wv, const float* bias_v,
                                              __bf16* vout) {
    __shared__ __bf16 Bt[64 * TROW_ELEMS];
    int wg = blockIdx.x;          // 2 * 98
    int b = wg / 98;
    int n0 = (wg % 98) * 64;
    int tid = threadIdx.x, wave = tid >> 5, lane = tid & 31;
    int c0 = wave * 32;           // 2 c-tiles per wave
    const __bf16* xrows = Xn + ((size_t)b * 6272 + n0) * 256;
    f32x8 acc[2][4] = {};
    for (int k0 = 0; k0 < 256; k0 += 32) {
        __syncthreads();
        async_stage_tile(xrows + k0, 256 * 2, Bt);
        __syncthreads();
        bf16x16 a0 = load_a_frag(wv + (size_t)c0 * 256 + k0, 256);
        bf16x16 a1 = load_a_frag(wv + (size_t)(c0 + 16) * 256 + k0, 256);
#pragma unroll
        for (int nt = 0; nt < 4; ++nt) {
            bf16x16 bf = load_b_frag(Bt + nt * 16 * TROW_ELEMS, TROW_ELEMS);
            acc[0][nt] = wmma_bf16(a0, bf, acc[0][nt]);
            acc[1][nt] = wmma_bf16(a1, bf, acc[1][nt]);
        }
    }
    __bf16* outp = vout + (size_t)b * 256 * 6272;
#pragma unroll
    for (int ct = 0; ct < 2; ++ct)
#pragma unroll
        for (int nt = 0; nt < 4; ++nt)
#pragma unroll
            for (int v = 0; v < 8; ++v) {
                int c = c0 + ct * 16 + v + 8 * ((lane >> 4) & 1);
                int n = n0 + nt * 16 + (lane & 15);
                float val = acc[ct][nt][v] + bias_v[c];
                outp[(size_t)c * 6272 + n] = (__bf16)fmaxf(val, 0.f);
            }
}

// ---------- kernel 4: column softmax stats (online) ----------
__global__ void __launch_bounds__(128) colstats(const __bf16* qT, const __bf16* kT,
                                                float* colmax, float* colsum) {
    int wg = blockIdx.x;          // 2 * 98
    int b = wg / 98;
    int m0 = (wg % 98) * 64 + (threadIdx.x >> 5) * 16;
    int lane = threadIdx.x & 31;
    const __bf16* qb = qT + (size_t)b * 6272 * 64;
    const __bf16* kb = kT + (size_t)b * 6272 * 64;
    bf16x16 kb0 = load_b_frag(kb + (size_t)m0 * 64, 64);
    bf16x16 kb1 = load_b_frag(kb + (size_t)m0 * 64 + 32, 64);
    float rmax = -1e30f, rsum = 0.f;
    for (int n0 = 0; n0 < 6272; n0 += 16) {
        bf16x16 a0 = load_a_frag(qb + (size_t)n0 * 64, 64);
        bf16x16 a1 = load_a_frag(qb + (size_t)n0 * 64 + 32, 64);
        f32x8 s = {};
        s = wmma_bf16(a0, kb0, s);
        s = wmma_bf16(a1, kb1, s);
        float tmax = s[0];
#pragma unroll
        for (int v = 1; v < 8; ++v) tmax = fmaxf(tmax, s[v]);
        tmax = fmaxf(tmax, __shfl_xor(tmax, 16, 32));
        float nmax = fmaxf(rmax, tmax);
        float scale = __expf(rmax - nmax);
        float ts = 0.f;
#pragma unroll
        for (int v = 0; v < 8; ++v) ts += __expf(s[v] - nmax);
        rsum = rsum * scale + ts;
        rmax = nmax;
    }
    rsum += __shfl_xor(rsum, 16, 32);
    if (lane < 16) {
        colmax[b * 6272 + m0 + lane] = rmax;
        colsum[b * 6272 + m0 + lane] = rsum;
    }
}

// ---------- kernel 5: vo[c][m] = (1/colsum) * sum_n v[c][n]*exp(s[n][m]-colmax) ----------
__global__ void __launch_bounds__(128) attn_apply(const __bf16* qT, const __bf16* kT, const __bf16* vmat,
                                                  const float* colmax, const float* colsum, float* vo) {
    __shared__ __bf16 P[16][66];  // [m_local][n within 64-chunk], padded
    int wg = blockIdx.x;          // 2 * 392
    int b = wg / 392;
    int m0 = (wg % 392) * 16;
    int tid = threadIdx.x, wave = tid >> 5, lane = tid & 31;
    const __bf16* qb = qT + (size_t)b * 6272 * 64;
    const __bf16* kb = kT + (size_t)b * 6272 * 64;
    const __bf16* vb = vmat + (size_t)b * 256 * 6272;
    int mcol = m0 + (lane & 15);
    float cmax = colmax[b * 6272 + mcol];
    bf16x16 kb0 = load_b_frag(kb + (size_t)m0 * 64, 64);
    bf16x16 kb1 = load_b_frag(kb + (size_t)m0 * 64 + 32, 64);
    int c0 = wave * 64;
    f32x8 acc[4] = {};
    for (int n0 = 0; n0 < 6272; n0 += 64) {
        int ns = n0 + wave * 16;
        bf16x16 a0 = load_a_frag(qb + (size_t)ns * 64, 64);
        bf16x16 a1 = load_a_frag(qb + (size_t)ns * 64 + 32, 64);
        f32x8 s = {};
        s = wmma_bf16(a0, kb0, s);
        s = wmma_bf16(a1, kb1, s);
        __syncthreads();  // previous chunk's P consumed
#pragma unroll
        for (int v = 0; v < 8; ++v) {
            int nl = wave * 16 + v + 8 * ((lane >> 4) & 1);
            P[lane & 15][nl] = (__bf16)__expf(s[v] - cmax);
        }
        __syncthreads();
#pragma unroll
        for (int ks = 0; ks < 2; ++ks) {
            bf16x16 pb = load_b_frag(&P[0][ks * 32], 66);
#pragma unroll
            for (int ct = 0; ct < 4; ++ct) {
                bf16x16 a = load_a_frag(vb + (size_t)(c0 + ct * 16) * 6272 + n0 + ks * 32, 6272);
                acc[ct] = wmma_bf16(a, pb, acc[ct]);
            }
        }
    }
    float inv = 1.f / colsum[b * 6272 + mcol];
    float* vob = vo + (size_t)b * 256 * 6272;
#pragma unroll
    for (int ct = 0; ct < 4; ++ct)
#pragma unroll
        for (int v = 0; v < 8; ++v) {
            int c = c0 + ct * 16 + v + 8 * ((lane >> 4) & 1);
            vob[(size_t)c * 6272 + mcol] = acc[ct][v] * inv;
        }
}

// ---------- kernel 6: out = relu(Wo' * On + bias_o) ----------
__global__ void __launch_bounds__(256) out_conv(const __bf16* On, const __bf16* wo,
                                                const float* bias_o, float* out) {
    __shared__ __bf16 Bt[64 * TROW_ELEMS];
    int wg = blockIdx.x;          // 196 = 12544/64 positions
    int pglob = wg * 64;
    int bb = pglob / 3136;
    int p0 = pglob % 3136;
    int tid = threadIdx.x, wave = tid >> 5, lane = tid & 31;
    int c0 = wave * 32;           // 2 oc-tiles per wave, 8 waves -> 256 oc
    const __bf16* orows = On + (size_t)pglob * 512;
    f32x8 acc[2][4] = {};
    for (int k0 = 0; k0 < 512; k0 += 32) {
        __syncthreads();
        async_stage_tile(orows + k0, 512 * 2, Bt);
        __syncthreads();
        bf16x16 a0 = load_a_frag(wo + (size_t)c0 * 512 + k0, 512);
        bf16x16 a1 = load_a_frag(wo + (size_t)(c0 + 16) * 512 + k0, 512);
#pragma unroll
        for (int nt = 0; nt < 4; ++nt) {
            bf16x16 bf = load_b_frag(Bt + nt * 16 * TROW_ELEMS, TROW_ELEMS);
            acc[0][nt] = wmma_bf16(a0, bf, acc[0][nt]);
            acc[1][nt] = wmma_bf16(a1, bf, acc[1][nt]);
        }
    }
    float* ob = out + (size_t)bb * 256 * 3136;
#pragma unroll
    for (int ct = 0; ct < 2; ++ct)
#pragma unroll
        for (int nt = 0; nt < 4; ++nt)
#pragma unroll
            for (int v = 0; v < 8; ++v) {
                int oc = c0 + ct * 16 + v + 8 * ((lane >> 4) & 1);
                int p = p0 + nt * 16 + (lane & 15);
                ob[(size_t)oc * 3136 + p] = fmaxf(acc[ct][nt][v] + bias_o[oc], 0.f);
            }
}

// ---------- launch ----------
extern "C" void kernel_launch(void* const* d_in, const int* in_sizes, int n_in,
                              void* d_out, int out_size, void* d_ws, size_t ws_size,
                              hipStream_t stream) {
    (void)in_sizes; (void)n_in; (void)out_size; (void)ws_size;
    const float* t  = (const float*)d_in[0];
    const float* Wq = (const float*)d_in[1];
    const float* Wk = (const float*)d_in[2];
    const float* Wv = (const float*)d_in[3];
    const float* gv = (const float*)d_in[4];
    const float* bv = (const float*)d_in[5];
    const float* mv = (const float*)d_in[6];
    const float* vv = (const float*)d_in[7];
    const float* Wo = (const float*)d_in[8];
    const float* go = (const float*)d_in[9];
    const float* bo = (const float*)d_in[10];
    const float* mo = (const float*)d_in[11];
    const float* vvo = (const float*)d_in[12];
    float* outp = (float*)d_out;

    char* ws = (char*)d_ws;
    size_t off = 0;
    auto alloc = [&](size_t bytes) -> void* {
        void* p = ws + off;
        off = (off + bytes + 255) & ~(size_t)255;
        return p;
    };
    __bf16* wq_bf = (__bf16*)alloc(64 * 256 * 2);
    __bf16* wk_bf = (__bf16*)alloc(64 * 256 * 2);
    __bf16* wv_bf = (__bf16*)alloc(256 * 256 * 2);
    __bf16* wo_bf = (__bf16*)alloc(256 * 512 * 2);
    float* bias_v = (float*)alloc(256 * 4);
    float* bias_o = (float*)alloc(256 * 4);
    __bf16* Xn   = (__bf16*)alloc((size_t)2 * 6272 * 256 * 2);
    __bf16* On   = (__bf16*)alloc((size_t)4 * 3136 * 512 * 2);
    __bf16* qT   = (__bf16*)alloc((size_t)2 * 6272 * 64 * 2);
    __bf16* kT   = (__bf16*)alloc((size_t)2 * 6272 * 64 * 2);
    __bf16* vbuf = (__bf16*)alloc((size_t)2 * 256 * 6272 * 2);
    float* cmax  = (float*)alloc((size_t)2 * 6272 * 4);
    float* csum  = (float*)alloc((size_t)2 * 6272 * 4);
    float* vobuf = (float*)alloc((size_t)2 * 256 * 6272 * 4);

    pack_weights<<<512, 256, 0, stream>>>(Wq, Wk, Wv, gv, bv, mv, vv, Wo, go, bo, mo, vvo,
                                          wq_bf, wk_bf, wv_bf, wo_bf, bias_v, bias_o);
    pack_x<<<12544, 256, 0, stream>>>(t, Xn);
    qk_proj<<<196, 256, 0, stream>>>(Xn, wq_bf, wk_bf, qT, kT);
    v_proj<<<196, 256, 0, stream>>>(Xn, wv_bf, bias_v, vbuf);
    colstats<<<196, 128, 0, stream>>>(qT, kT, cmax, csum);
    attn_apply<<<784, 128, 0, stream>>>(qT, kT, vbuf, cmax, csum, vobuf);
    pack_o<<<25088, 256, 0, stream>>>(t, vobuf, On);
    out_conv<<<196, 256, 0, stream>>>(On, wo_bf, bias_o, outp);
}
